// JRTransformer_46566035423952
// MI455X (gfx1250) — matrix operand order repaired and checked by reading, not compiled, not run on hardware
//
#include <hip/hip_runtime.h>

typedef __attribute__((ext_vector_type(16))) _Float16 v16h;
typedef __attribute__((ext_vector_type(8)))  _Float16 v8h;
typedef __attribute__((ext_vector_type(8)))  float    v8f;

#define WMMA_F16(a, b, c) \
  __builtin_amdgcn_wmma_f32_16x16x32_f16(false, (a), false, (b), (short)0, (c), false, false)

namespace {
constexpr int   kDepth  = 6;
constexpr int   kN      = 15;     // joints
constexpr int   kDim    = 128;
constexpr int   kHeads  = 16;
constexpr int   kHid    = 64;     // mlp hidden
constexpr float kScale  = 0.6f;
constexpr float kEps    = 1e-5f;

constexpr int kFragH   = 512;     // halves per 16x32 B fragment (32 lanes * 16)
constexpr int kJFrags  = 24 * 4;  // 128x384 GEMM: 24 col tiles, 4 k-chunks
constexpr int kPFrags  = 8 * 4;   // 128x128
constexpr int kF1Frags = 4 * 4;   // 128x64
constexpr int kF2Frags = 8 * 2;   // 64x128
constexpr int kConvH   = 128;     // CwT: 16 padded joints * 8 (halves)
}

// ---------------------------------------------------------------------------
// Weight prep: fp32 row-major [K x Ncol] -> f16 WMMA B-fragment layout.
// Fragment f = ct*KC + kc.  Lane L holds column n = ct*16 + (L&15) and
// K rows kc*32 + (L>>4)*16 + h, h=0..15 (32 contiguous bytes per lane).
// ---------------------------------------------------------------------------
__global__ void prep_frag(const float* __restrict__ W, _Float16* __restrict__ out,
                          int Ncol, int KC) {
  const int layer = blockIdx.y;
  const int f     = blockIdx.x;
  const int ct    = f / KC;
  const int kc    = f % KC;
  const int lane  = threadIdx.x;
  const int K     = KC * 32;
  const float* Wl = W + (size_t)layer * (size_t)K * (size_t)Ncol;
  _Float16*  ol   = out + ((size_t)layer * gridDim.x + f) * kFragH + lane * 16;
  const int col   = ct * 16 + (lane & 15);
  const int krow  = kc * 32 + (lane >> 4) * 16;
#pragma unroll
  for (int h = 0; h < 16; ++h)
    ol[h] = (_Float16)Wl[(size_t)(krow + h) * Ncol + col];
}

// Iconv_w [HS=8 x N=15] -> CwT[n][d] f16, n padded to 16 with zeros.
__global__ void prep_conv(const float* __restrict__ W, _Float16* __restrict__ out) {
  const int layer = blockIdx.x;
  const int t = threadIdx.x;       // 128 threads
  const int n = t >> 3, d = t & 7;
  out[(size_t)layer * kConvH + n * 8 + d] =
      (n < kN) ? (_Float16)W[layer * (8 * kN) + d * kN + n] : (_Float16)0.0f;
}

// ---------------------------------------------------------------------------
// Main kernel: one wave32 per batch element.
// ---------------------------------------------------------------------------
struct Params {
  const float* joint;
  const float* rel;
  const float *Jb, *Ib, *Cb, *Pb;
  const float *l1w, *l1b, *l2w, *l2b, *l3w, *l3b;
  const float *f1b, *f2b;
  const _Float16 *wJ, *wI, *wP, *wF1, *wF2, *wC;
  float* out;
  int batch;
};

// LayerNorm over DIM=128: lane handles 4 contiguous features of each row.
// Writes f16 rows 0..14 into dst (16 x 128), zeros row 15 (M padding).
__device__ inline void ln_rows(const float* __restrict__ src,
                               const float* __restrict__ w,
                               const float* __restrict__ bb,
                               _Float16* dst, int lane) {
  const int c = lane * 4;
  const float w0 = w[c], w1 = w[c + 1], w2 = w[c + 2], w3 = w[c + 3];
  const float b0 = bb[c], b1 = bb[c + 1], b2 = bb[c + 2], b3 = bb[c + 3];
  for (int r = 0; r < kN; ++r) {
    const float* s = src + r * kDim + c;
    float x0 = s[0], x1 = s[1], x2 = s[2], x3 = s[3];
    float sm = x0 + x1 + x2 + x3;
    float sq = x0 * x0 + x1 * x1 + x2 * x2 + x3 * x3;
#pragma unroll
    for (int m = 1; m < 32; m <<= 1) {
      sm += __shfl_xor(sm, m, 32);
      sq += __shfl_xor(sq, m, 32);
    }
    const float mean = sm * (1.0f / kDim);
    const float rstd = rsqrtf(sq * (1.0f / kDim) - mean * mean + kEps);
    _Float16* d = dst + r * kDim + c;
    d[0] = (_Float16)((x0 - mean) * rstd * w0 + b0);
    d[1] = (_Float16)((x1 - mean) * rstd * w1 + b1);
    d[2] = (_Float16)((x2 - mean) * rstd * w2 + b2);
    d[3] = (_Float16)((x3 - mean) * rstd * w3 + b3);
  }
  _Float16* d = dst + 15 * kDim + c;
  d[0] = (_Float16)0.0f; d[1] = (_Float16)0.0f;
  d[2] = (_Float16)0.0f; d[3] = (_Float16)0.0f;
}

// A fragment (16x32 f16) from LDS row-major buffer.
// Lane L: row = L&15; lanes 0-15 carry K {0..15}, lanes 16-31 {16..31}.
__device__ inline v16h load_afrag(const _Float16* A, int lda, int row, int kbase) {
  const v8h lo8 = *(const v8h*)(A + row * lda + kbase);
  const v8h hi8 = *(const v8h*)(A + row * lda + kbase + 8);
  v16h a;
#pragma unroll
  for (int j = 0; j < 8; ++j) { a[j] = lo8[j]; a[8 + j] = hi8[j]; }
  return a;
}

// Generic tiled GEMM: D(16 x NT*16) = A(16 x KC*32) * Wfrag + bias, epilogue by MODE.
// MODE 0: store f16 row-major to out (ldo).
// MODE 1: residual accumulate into xres f32 (rows < 15 only).
// MODE 2: exact GELU then store f16 row-major to out.
// MODE 3: store f16 column-major [col][16 rows] to out (single b128 per lane/tile).
template <int MODE>
__device__ inline void gemm(const _Float16* A, int lda,
                            const _Float16* __restrict__ Wf,
                            const float* __restrict__ bias,
                            int NT, int KC,
                            _Float16* out, int ldo,
                            float* xres, int lane) {
  const int lo = lane & 15, hi = lane >> 4;
  for (int ct = 0; ct < NT; ++ct) {
    v8f acc = {};
#pragma unroll 4
    for (int kc = 0; kc < KC; ++kc) {
      const v16h a = load_afrag(A, lda, lo, kc * 32 + hi * 16);
      const v16h b = *(const v16h*)(Wf + (size_t)(ct * KC + kc) * kFragH + lane * 16);
      acc = WMMA_F16(a, b, acc);
    }
    const int col = ct * 16 + lo;
    const float bv = bias[col];
    if (MODE == 3) {
      v8h pk;
#pragma unroll
      for (int r = 0; r < 8; ++r) pk[r] = (_Float16)(acc[r] + bv);
      // D rows m = hi*8 + r are contiguous in [col][row] layout -> 1x ds_store_b128
      *(v8h*)(out + col * 16 + hi * 8) = pk;
    } else {
#pragma unroll
      for (int r = 0; r < 8; ++r) {
        const int m = r + hi * 8;  // D layout: lanes 0-15 -> M=r, lanes 16-31 -> M=r+8
        const float v = acc[r] + bv;
        if (MODE == 0) {
          out[m * ldo + col] = (_Float16)v;
        } else if (MODE == 1) {
          if (m < kN) xres[m * kDim + col] += v;
        } else {
          const float g = 0.5f * v * (1.0f + erff(v * 0.70710678118f));
          out[m * ldo + col] = (_Float16)g;
        }
      }
    }
  }
}

__global__ __launch_bounds__(32) void jr_main(Params p) {
  __shared__ alignas(32) float    xf[kN * kDim];       // residual stream (f32)
  __shared__ alignas(32) _Float16 aj[16 * kDim];       // LN1 / LN3 output (A operand)
  __shared__ alignas(32) _Float16 arho[16 * kDim];     // LN2 output, then attention out
  __shared__ alignas(32) _Float16 qkvJ[16 * 256];      // Jq | Jk (row-major)
  __shared__ alignas(32) _Float16 qkvI[16 * 384];      // Iq | Ik | Iv (row-major)
  __shared__ alignas(32) _Float16 vT[kDim * 16];       // Jv transposed: [col][joint]
  __shared__ alignas(32) _Float16 probs[16 * 16];
  __shared__ alignas(32) _Float16 gbuf[16 * kHid];

  const int lane = threadIdx.x;
  const int lo = lane & 15, hi = lane >> 4;
  const int b = blockIdx.x;
  if (b >= p.batch) return;

  const float* jp = p.joint + (size_t)b * kN * kDim;
  const float* rp = p.rel + (size_t)b * kN * kDim;
  for (int i = lane; i < kN * kDim; i += 32) xf[i] = jp[i];

  for (int L = 0; L < kDepth; ++L) {
    const _Float16* wJ  = p.wJ  + (size_t)L * (kJFrags  * kFragH);
    const _Float16* wI  = p.wI  + (size_t)L * (kJFrags  * kFragH);
    const _Float16* wP  = p.wP  + (size_t)L * (kPFrags  * kFragH);
    const _Float16* wF1 = p.wF1 + (size_t)L * (kF1Frags * kFragH);
    const _Float16* wF2 = p.wF2 + (size_t)L * (kF2Frags * kFragH);
    const _Float16* cT  = p.wC  + (size_t)L * kConvH;

    // --- LN1(x) -> aj, LN2(relation) -> arho ---------------------------------
    ln_rows(xf, p.l1w + L * kDim, p.l1b + L * kDim, aj, lane);
    ln_rows(rp, p.l2w + L * kDim, p.l2b + L * kDim, arho, lane);

    // --- QKV GEMMs (15x128 @ 128x384) ----------------------------------------
    // Jq,Jk row-major (tiles 0..15); Jv straight into transposed vT (tiles 16..23)
    gemm<0>(aj, kDim, wJ, p.Jb + L * 384, 16, 4, qkvJ, 256, nullptr, lane);
    gemm<3>(aj, kDim, wJ + (size_t)(16 * 4) * kFragH, p.Jb + L * 384 + 256,
            8, 4, vT, 0, nullptr, lane);
    gemm<0>(arho, kDim, wI, p.Ib + L * 384, 24, 4, qkvI, 384, nullptr, lane);

    // --- attention: per head, one WMMA computes JqJk^T + IqIk^T + Iv@Cw ------
    const float cb = (lo < kN) ? p.Cb[L * kN + lo] : 0.0f;
    for (int h = 0; h < kHeads; ++h) {
      const int qo = h * 8;
      v16h af, bf;
      if (hi == 0) {  // K 0..7 = Jq, K 8..15 = Iq   |   B: K 0..7 = Jk, 8..15 = Ik
        const v8h q  = *(const v8h*)(qkvJ + lo * 256 + qo);
        const v8h iv = *(const v8h*)(qkvI + lo * 384 + qo);
        const v8h kj = *(const v8h*)(qkvJ + lo * 256 + 128 + qo);
        const v8h ki = *(const v8h*)(qkvI + lo * 384 + 128 + qo);
#pragma unroll
        for (int j = 0; j < 8; ++j) {
          af[j] = q[j];  af[8 + j] = iv[j];
          bf[j] = kj[j]; bf[8 + j] = ki[j];
        }
      } else {        // K 16..23 = Iv, K 24..31 = 0   |   B: K 16..23 = CwT, 24..31 = 0
        const v8h iq = *(const v8h*)(qkvI + lo * 384 + 256 + qo);
        const v8h cw = *(const v8h*)(cT + lo * 8);
#pragma unroll
        for (int j = 0; j < 8; ++j) {
          af[j] = iq[j]; af[8 + j] = (_Float16)0.0f;
          bf[j] = cw[j]; bf[8 + j] = (_Float16)0.0f;
        }
      }
      v8f sc = {};
      sc = WMMA_F16(af, bf, sc);

      // row-wise softmax over 15 valid key columns (16-lane half-wave groups)
#pragma unroll
      for (int r = 0; r < 8; ++r) {
        const float val = (lo < kN) ? (sc[r] + cb) * kScale : -1e30f;
        float mx = val;
#pragma unroll
        for (int m = 1; m < 16; m <<= 1) mx = fmaxf(mx, __shfl_xor(mx, m, 32));
        const float e = (lo < kN) ? __expf(val - mx) : 0.0f;
        float sum = e;
#pragma unroll
        for (int m = 1; m < 16; m <<= 1) sum += __shfl_xor(sum, m, 32);
        probs[(r + hi * 8) * 16 + lo] = (_Float16)(e / sum);
      }

      // attn (16x16, K=joints) @ Jv (joints x 8)
      v16h pa, vb;
      if (hi == 0) {
        const v16h p0 = *(const v16h*)(probs + lo * 16);
#pragma unroll
        for (int j = 0; j < 16; ++j) pa[j] = p0[j];
        vb = *(const v16h*)(vT + (qo + (lo & 7)) * 16);
      } else {
#pragma unroll
        for (int j = 0; j < 16; ++j) pa[j] = (_Float16)0.0f;
#pragma unroll
        for (int j = 0; j < 16; ++j) vb[j] = (_Float16)0.0f;
      }
      v8f oa = {};
      oa = WMMA_F16(pa, vb, oa);
      if (lo < 8) {
#pragma unroll
        for (int r = 0; r < 8; ++r)
          arho[(r + hi * 8) * kDim + qo + lo] = (_Float16)oa[r];
      }
    }

    // --- proj + residual ------------------------------------------------------
    gemm<1>(arho, kDim, wP, p.Pb + L * kDim, 8, 4, nullptr, 0, xf, lane);

    // --- MLP: LN3 -> fc1+GELU -> fc2 + residual -------------------------------
    ln_rows(xf, p.l3w + L * kDim, p.l3b + L * kDim, aj, lane);
    gemm<2>(aj, kDim, wF1, p.f1b + L * kHid, 4, 4, gbuf, kHid, nullptr, lane);
    gemm<1>(gbuf, kHid, wF2, p.f2b + L * kDim, 8, 2, nullptr, 0, xf, lane);
  }

  float* op = p.out + (size_t)b * kN * kDim;
  for (int i = lane; i < kN * kDim; i += 32) op[i] = xf[i];
}

// ---------------------------------------------------------------------------
extern "C" void kernel_launch(void* const* d_in, const int* in_sizes, int n_in,
                              void* d_out, int out_size, void* d_ws, size_t ws_size,
                              hipStream_t stream) {
  const float* joint = (const float*)d_in[0];
  const float* rel   = (const float*)d_in[1];
  const float* Jw    = (const float*)d_in[2];
  const float* Jb    = (const float*)d_in[3];
  const float* Iw    = (const float*)d_in[4];
  const float* Ib    = (const float*)d_in[5];
  const float* Cw    = (const float*)d_in[6];
  const float* Cb    = (const float*)d_in[7];
  const float* Pw    = (const float*)d_in[8];
  const float* Pb    = (const float*)d_in[9];
  const float* l1w   = (const float*)d_in[10];
  const float* l1b   = (const float*)d_in[11];
  const float* l2w   = (const float*)d_in[12];
  const float* l2b   = (const float*)d_in[13];
  const float* l3w   = (const float*)d_in[14];
  const float* l3b   = (const float*)d_in[15];
  const float* F1w   = (const float*)d_in[16];
  const float* F1b   = (const float*)d_in[17];
  const float* F2w   = (const float*)d_in[18];
  const float* F2b   = (const float*)d_in[19];

  const int batch = in_sizes[0] / (kN * kDim);

  // workspace layout (f16 fragment-packed weights)
  _Float16* ws  = (_Float16*)d_ws;
  _Float16* wsJ  = ws;
  _Float16* wsI  = wsJ  + (size_t)kDepth * kJFrags  * kFragH;
  _Float16* wsP  = wsI  + (size_t)kDepth * kJFrags  * kFragH;
  _Float16* wsF1 = wsP  + (size_t)kDepth * kPFrags  * kFragH;
  _Float16* wsF2 = wsF1 + (size_t)kDepth * kF1Frags * kFragH;
  _Float16* wsC  = wsF2 + (size_t)kDepth * kF2Frags * kFragH;

  prep_frag<<<dim3(kJFrags,  kDepth), 32, 0, stream>>>(Jw,  wsJ,  384, 4);
  prep_frag<<<dim3(kJFrags,  kDepth), 32, 0, stream>>>(Iw,  wsI,  384, 4);
  prep_frag<<<dim3(kPFrags,  kDepth), 32, 0, stream>>>(Pw,  wsP,  128, 4);
  prep_frag<<<dim3(kF1Frags, kDepth), 32, 0, stream>>>(F1w, wsF1,  64, 4);
  prep_frag<<<dim3(kF2Frags, kDepth), 32, 0, stream>>>(F2w, wsF2, 128, 2);
  prep_conv<<<kDepth, 128, 0, stream>>>(Cw, wsC);

  Params p;
  p.joint = joint; p.rel = rel;
  p.Jb = Jb; p.Ib = Ib; p.Cb = Cb; p.Pb = Pb;
  p.l1w = l1w; p.l1b = l1b; p.l2w = l2w; p.l2b = l2b; p.l3w = l3w; p.l3b = l3b;
  p.f1b = F1b; p.f2b = F2b;
  p.wJ = wsJ; p.wI = wsI; p.wP = wsP; p.wF1 = wsF1; p.wF2 = wsF2; p.wC = wsC;
  p.out = (float*)d_out;
  p.batch = batch;

  jr_main<<<batch, 32, 0, stream>>>(p);
}
